// GroupedQueryAttention_62551903699488
// MI455X (gfx1250) — matrix operand order repaired and checked by reading, not compiled, not run on hardware
//
#include <hip/hip_runtime.h>

typedef __attribute__((ext_vector_type(16))) _Float16 v16h;
typedef __attribute__((ext_vector_type(8)))  _Float16 v8h;
typedef __attribute__((ext_vector_type(4)))  _Float16 v4h;
typedef __attribute__((ext_vector_type(8)))  float    v8f;
typedef __attribute__((ext_vector_type(4)))  float    v4f;

#define B_   2
#define T_   2048
#define C_   4096
#define M_   4096   /* B_*T_ */
#define KV_  1024
#define HQ_  32
#define D_   128

// ---------------------------------------------------------------------------
// CDNA5 async global->LDS copy (ISA 07_vmem op 98 / 08_async_tensor §4).
// LDS operand = low 32 bits of the generic pointer (LDS aperture keeps the
// wave-relative offset in addr[31:0]).  Tracked by ASYNCcnt.
// ---------------------------------------------------------------------------
static __device__ __forceinline__ void async_b128(_Float16* lds,
                                                  const _Float16* g) {
  unsigned int l = (unsigned int)(unsigned long long)(uintptr_t)lds;
  unsigned long long ga = (unsigned long long)(uintptr_t)g;
  asm volatile("global_load_async_to_lds_b128 %0, %1, off"
               :: "v"(l), "v"(ga) : "memory");
}

template <int N>
static __device__ __forceinline__ void wait_async() {
  asm volatile("s_wait_asynccnt %0" :: "n"(N) : "memory");
}

// ---------------------------------------------------------------------------
// WMMA helpers (gfx1250 v_wmma_f32_16x16x32_f16)
// ---------------------------------------------------------------------------
static __device__ __forceinline__ v8f wmma16(v16h a, v16h b, v8f c) {
  return __builtin_amdgcn_wmma_f32_16x16x32_f16(
      false, a, false, b, (short)0, c, false, false);
}

// A-matrix fragment (16x32 f16): lanes 0-15 -> row M=lane, K = 0..15;
// lanes 16-31 -> row M=lane-16, K = 16..31.  Two 16B loads per lane.
static __device__ __forceinline__ v16h load_a_frag(const _Float16* p, int ld) {
  const int lane = threadIdx.x & 31;
  const int m  = lane & 15;
  const int kh = (lane >> 4) << 4;    // 0 or 16
  const _Float16* q = p + (size_t)m * ld + kh;
  v8h lo = *(const v8h*)(q);
  v8h hi = *(const v8h*)(q + 8);
  v16h r;
#pragma unroll
  for (int i = 0; i < 8; ++i) { r[i] = lo[i]; r[8 + i] = hi[i]; }
  return r;
}

// B-matrix fragment (32x16 f16): lane holds column N=lane&15; lanes 0-15 hold
// K=0..15, lanes 16-31 K=16..31.  LDS flavor: two 16B loads (row strides are
// only 16B-aligned).
static __device__ __forceinline__ v16h load_b_frag_lds(const _Float16* p,
                                                       int ld) {
  const int lane = threadIdx.x & 31;
  const int n  = lane & 15;
  const int kb = (lane >> 4) << 4;    // 0 or 16
  const _Float16* q = p + (size_t)n * ld + kb;
  v8h lo = *(const v8h*)(q);
  v8h hi = *(const v8h*)(q + 8);
  v16h r;
#pragma unroll
  for (int i = 0; i < 8; ++i) { r[i] = lo[i]; r[8 + i] = hi[i]; }
  return r;
}

// Global flavor: one 32B load per lane (addresses are 32B-aligned).
static __device__ __forceinline__ v16h load_b_frag_g(const _Float16* p,
                                                     int ld) {
  const int lane = threadIdx.x & 31;
  const int n  = lane & 15;
  const int kb = (lane >> 4) << 4;    // 0 or 16
  return *(const v16h*)(p + (size_t)n * ld + kb);
}

// ---------------------------------------------------------------------------
// fp32 -> f16 conversion (4-wide)
// ---------------------------------------------------------------------------
__global__ void cvt_kernel(const float* __restrict__ in,
                           _Float16* __restrict__ out, int n4) {
  int i = blockIdx.x * blockDim.x + threadIdx.x;
  if (i < n4) {
    v4f v = *(const v4f*)(in + (size_t)i * 4);
    v4h h;
#pragma unroll
    for (int j = 0; j < 4; ++j) h[j] = (_Float16)v[j];
    *(v4h*)(out + (size_t)i * 4) = h;
  }
}

// ---------------------------------------------------------------------------
// GEMM:  Y[M,N] = X[M,K] @ W[N,K]^T, async-LDS double-buffered.
// Block 256 thr = 8 waves (4x2); block tile 128x128; k-step 32.
// LDS tiles 128 rows x 32 halves, row stride 40 halves (80B: 16B-aligned,
// conflict-free on 64 banks).  4 async b128 per thread per k-step.
// MODE 0: f32 row-major   MODE 1: f16 row-major   MODE 2: f16 transposed
// ---------------------------------------------------------------------------
static __device__ __forceinline__ void stage_gemm(
    const _Float16* __restrict__ A, const _Float16* __restrict__ W,
    _Float16* Ab, _Float16* Bb, int row0, int col0, int K, int k0) {
  const int tid = threadIdx.x;
#pragma unroll
  for (int c = 0; c < 2; ++c) {
    const int ch  = tid + c * 256;       // 0..511
    const int row = ch >> 2;             // 0..127
    const int sub = ch & 3;              // 16B chunk within 64B row
    async_b128(Ab + row * 40 + sub * 8,
               A + (size_t)(row0 + row) * K + k0 + sub * 8);
    async_b128(Bb + row * 40 + sub * 8,
               W + (size_t)(col0 + row) * K + k0 + sub * 8);
  }
}

template <int MODE>
__global__ __launch_bounds__(256) void gemm_xwt_kernel(
    const _Float16* __restrict__ A, const _Float16* __restrict__ W,
    void* __restrict__ Cp, int M, int N, int K) {
  const int wave = threadIdx.x >> 5;
  const int lane = threadIdx.x & 31;
  const int wrow = (wave >> 1) * 32;    // 0,32,64,96 (local)
  const int wcol = (wave & 1) * 64;     // 0,64      (local)
  const int row0 = blockIdx.y * 128;
  const int col0 = blockIdx.x * 128;

  __shared__ __attribute__((aligned(16))) _Float16 sA[2][128 * 40];
  __shared__ __attribute__((aligned(16))) _Float16 sB[2][128 * 40];

  v8f acc[2][4];
#pragma unroll
  for (int i = 0; i < 2; ++i)
#pragma unroll
    for (int n = 0; n < 4; ++n) { v8f z = {}; acc[i][n] = z; }

  const int KT = K >> 5;
  stage_gemm(A, W, sA[0], sB[0], row0, col0, K, 0);
  stage_gemm(A, W, sA[1], sB[1], row0, col0, K, 32);

  for (int kt = 0; kt < KT; ++kt) {
    if (kt + 1 < KT) wait_async<4>(); else wait_async<0>();
    __syncthreads();
    const _Float16* Ab = sA[kt & 1];
    const _Float16* Bb = sB[kt & 1];
    v16h a0 = load_a_frag(Ab + (wrow +  0) * 40, 40);
    v16h a1 = load_a_frag(Ab + (wrow + 16) * 40, 40);
    v16h b0 = load_b_frag_lds(Bb + (wcol +  0) * 40, 40);
    v16h b1 = load_b_frag_lds(Bb + (wcol + 16) * 40, 40);
    v16h b2 = load_b_frag_lds(Bb + (wcol + 32) * 40, 40);
    v16h b3 = load_b_frag_lds(Bb + (wcol + 48) * 40, 40);
    acc[0][0] = wmma16(a0, b0, acc[0][0]);
    acc[0][1] = wmma16(a0, b1, acc[0][1]);
    acc[0][2] = wmma16(a0, b2, acc[0][2]);
    acc[0][3] = wmma16(a0, b3, acc[0][3]);
    acc[1][0] = wmma16(a1, b0, acc[1][0]);
    acc[1][1] = wmma16(a1, b1, acc[1][1]);
    acc[1][2] = wmma16(a1, b2, acc[1][2]);
    acc[1][3] = wmma16(a1, b3, acc[1][3]);
    __syncthreads();
    if (kt + 2 < KT)
      stage_gemm(A, W, sA[kt & 1], sB[kt & 1], row0, col0, K, (kt + 2) * 32);
  }

  // C/D layout: VGPR i -> (row = i + (lane>=16 ? 8 : 0), col = lane&15)
  const int cn = lane & 15;
  const int rb = (lane >> 4) << 3;
  if (MODE == 0) {
    float* Cf = (float*)Cp;
#pragma unroll
    for (int it = 0; it < 2; ++it)
#pragma unroll
      for (int nt = 0; nt < 4; ++nt)
#pragma unroll
        for (int i = 0; i < 8; ++i)
          Cf[(size_t)(row0 + wrow + it * 16 + rb + i) * N +
             (col0 + wcol + nt * 16 + cn)] = acc[it][nt][i];
  } else if (MODE == 1) {
    _Float16* Ch = (_Float16*)Cp;
#pragma unroll
    for (int it = 0; it < 2; ++it)
#pragma unroll
      for (int nt = 0; nt < 4; ++nt)
#pragma unroll
        for (int i = 0; i < 8; ++i)
          Ch[(size_t)(row0 + wrow + it * 16 + rb + i) * N +
             (col0 + wcol + nt * 16 + cn)] = (_Float16)acc[it][nt][i];
  } else {
    _Float16* Ct = (_Float16*)Cp;   // Y_t[n*M + m]; 8 rows contiguous per lane
#pragma unroll
    for (int it = 0; it < 2; ++it)
#pragma unroll
      for (int nt = 0; nt < 4; ++nt) {
        v8h v;
#pragma unroll
        for (int i = 0; i < 8; ++i) v[i] = (_Float16)acc[it][nt][i];
        *(v8h*)(Ct + (size_t)(col0 + wcol + nt * 16 + cn) * M +
                row0 + wrow + it * 16 + rb) = v;
      }
  }
}

// ---------------------------------------------------------------------------
// Flash-style GQA attention, S computed TRANSPOSED: S^T = K_tile @ Q^T.
// Each lane then owns one query column -> softmax is an in-lane 16-value
// reduction + a single xor-16 shuffle (instead of 64 ds_bpermute per step).
// K/V tiles async-staged to LDS, shared by all 8 waves, double-buffered.
// grid (T/128, HQ, B); block 256; each wave owns 16 query rows.
// ---------------------------------------------------------------------------
static __device__ __forceinline__ void stage_kv(
    const _Float16* __restrict__ Kbase, const _Float16* __restrict__ Vbase,
    _Float16* kb, _Float16* vb, int j0) {
  const int tid = threadIdx.x;
  {
    const int row = tid >> 3;            // 0..31 (key)
    const int sub = tid & 7;             // 16B chunk within 256B row
    async_b128(kb + row * 136 + sub * 8,
               Kbase + (size_t)(j0 + row) * KV_ + sub * 8);
  }
#pragma unroll
  for (int c = 0; c < 2; ++c) {
    const int ch  = tid + c * 256;       // 0..511
    const int row = ch >> 2;             // 0..127 (d)
    const int sub = ch & 3;              // 16B chunk within 64B row
    async_b128(vb + row * 40 + sub * 8,
               Vbase + (size_t)row * M_ + j0 + sub * 8);
  }
}

__global__ __launch_bounds__(256) void attn_kernel(
    const _Float16* __restrict__ Qh, const _Float16* __restrict__ Kh,
    const _Float16* __restrict__ Vt, _Float16* __restrict__ Ah) {
  const int wave = threadIdx.x >> 5;
  const int lane = threadIdx.x & 31;
  const int h  = blockIdx.y;
  const int b  = blockIdx.z;
  const int hk = h >> 2;                       // n_rep = 4
  const int q0 = blockIdx.x * 128 + wave * 16;

  __shared__ __attribute__((aligned(16))) _Float16 sK[2][32 * 136];
  __shared__ __attribute__((aligned(16))) _Float16 sV[2][128 * 40];
  __shared__ __attribute__((aligned(16))) _Float16 sP[8][16 * 40];
  _Float16* myp = sP[wave];

  const _Float16* Qbase = Qh + ((size_t)(b * T_ + q0)) * C_ + h * D_;
  const _Float16* Kbase = Kh + ((size_t)b * T_) * KV_ + hk * D_;
  const _Float16* Vbase = Vt + ((size_t)(hk * D_)) * M_ + b * T_;

  // Loop-invariant Q^T B-fragments: lane n holds query q0+n, d contiguous.
  v16h qfb[4];
#pragma unroll
  for (int c = 0; c < 4; ++c) qfb[c] = load_b_frag_g(Qbase + c * 32, C_);

  v8f o[8];
#pragma unroll
  for (int t = 0; t < 8; ++t) { v8f z = {}; o[t] = z; }

  float mq = -3.0e38f;   // running max for this lane's query (lane&15)
  float lq = 0.0f;       // running sum

  const float scale = 0.08838834764831845f;   // 1/sqrt(128)
  const int cn = lane & 15;
  const int rb = (lane >> 4) << 3;

  stage_kv(Kbase, Vbase, sK[0], sV[0], 0);
  stage_kv(Kbase, Vbase, sK[1], sV[1], 32);

  const int JT = T_ / 32;   // 64
  for (int jt = 0; jt < JT; ++jt) {
    if (jt + 1 < JT) wait_async<3>(); else wait_async<0>();
    __syncthreads();
    const _Float16* Kb = sK[jt & 1];
    const _Float16* Vb = sV[jt & 1];

    // S^T tiles: s0 = keys 0..15, s1 = keys 16..31 (rows), queries in cols.
    // A = K tile rows (keys) from LDS; B = loop-invariant Q^T frags.
    v8f s0 = {}, s1 = {};
    {
      v16h kf[4];
#pragma unroll
      for (int c = 0; c < 4; ++c) kf[c] = load_a_frag(Kb + c * 32, 136);
#pragma unroll
      for (int c = 0; c < 4; ++c) s0 = wmma16(kf[c], qfb[c], s0);
#pragma unroll
      for (int c = 0; c < 4; ++c)
        kf[c] = load_a_frag(Kb + 16 * 136 + c * 32, 136);
#pragma unroll
      for (int c = 0; c < 4; ++c) s1 = wmma16(kf[c], qfb[c], s1);
    }

    // lane holds 16 scores of query cn: keys rb+0..7 and 16+rb+0..7
    float sv[16];
#pragma unroll
    for (int i = 0; i < 8; ++i) {
      sv[i]     = s0[i] * scale;
      sv[8 + i] = s1[i] * scale;
    }
    float mloc = sv[0];
#pragma unroll
    for (int i = 1; i < 16; ++i) mloc = fmaxf(mloc, sv[i]);
    mloc = fmaxf(mloc, __shfl_xor(mloc, 16, 32));   // other 16 keys
    float mnew = fmaxf(mq, mloc);
    float al = __expf(mq - mnew);
    mq = mnew;
    float pv[16];
    float rs = 0.0f;
#pragma unroll
    for (int i = 0; i < 16; ++i) { pv[i] = __expf(sv[i] - mnew); rs += pv[i]; }
    rs += __shfl_xor(rs, 16, 32);
    lq = lq * al + rs;

    // P^T (C-layout) -> P row-major in per-wave LDS: lane writes query-row cn,
    // keys rb..rb+7 and 16+rb..16+rb+7 -> two contiguous b128 stores.
    v8h ph0, ph1;
#pragma unroll
    for (int i = 0; i < 8; ++i) {
      ph0[i] = (_Float16)pv[i];
      ph1[i] = (_Float16)pv[8 + i];
    }
    *(v8h*)(myp + cn * 40 + rb)      = ph0;
    *(v8h*)(myp + cn * 40 + 16 + rb) = ph1;

    // rescale O: row i of this lane is query rb+i; alpha lives in lane rb+i
    float ab[8];
#pragma unroll
    for (int i = 0; i < 8; ++i) ab[i] = __shfl(al, rb + i, 32);
#pragma unroll
    for (int t = 0; t < 8; ++t)
#pragma unroll
      for (int i = 0; i < 8; ++i) o[t][i] *= ab[i];

    // O += P @ V  (A = P from LDS, B = V chunks; grouped loads then WMMAs)
    v16h pf = load_a_frag(myp, 40);
#pragma unroll
    for (int g = 0; g < 2; ++g) {
      v16h vf[4];
#pragma unroll
      for (int t = 0; t < 4; ++t)
        vf[t] = load_b_frag_lds(Vb + (g * 4 + t) * 16 * 40, 40);
#pragma unroll
      for (int t = 0; t < 4; ++t)
        o[g * 4 + t] = wmma16(pf, vf[t], o[g * 4 + t]);
    }

    __syncthreads();
    if (jt + 2 < JT)
      stage_kv(Kbase, Vbase, sK[jt & 1], sV[jt & 1], (jt + 2) * 32);
  }

  // normalize: 1/l for query rb+i lives in lane rb+i
  float linv = 1.0f / lq;
  float lb[8];
#pragma unroll
  for (int i = 0; i < 8; ++i) lb[i] = __shfl(linv, rb + i, 32);

  _Float16* Ob = Ah + ((size_t)(b * T_ + q0)) * C_ + h * D_;
#pragma unroll
  for (int t = 0; t < 8; ++t)
#pragma unroll
    for (int i = 0; i < 8; ++i)
      Ob[(size_t)(rb + i) * C_ + t * 16 + cn] = (_Float16)(o[t][i] * lb[i]);
}

// ---------------------------------------------------------------------------
// Launch
// ---------------------------------------------------------------------------
extern "C" void kernel_launch(void* const* d_in, const int* in_sizes, int n_in,
                              void* d_out, int out_size, void* d_ws,
                              size_t ws_size, hipStream_t stream) {
  (void)in_sizes; (void)n_in; (void)out_size; (void)ws_size;
  const float* x  = (const float*)d_in[0];
  const float* Wq = (const float*)d_in[1];
  const float* Wk = (const float*)d_in[2];
  const float* Wv = (const float*)d_in[3];
  const float* Wo = (const float*)d_in[4];
  float* out = (float*)d_out;

  _Float16* ws = (_Float16*)d_ws;
  size_t off = 0;
  _Float16* xh  = ws + off; off += (size_t)M_  * C_;
  _Float16* Wqh = ws + off; off += (size_t)C_  * C_;
  _Float16* Wkh = ws + off; off += (size_t)KV_ * C_;
  _Float16* Wvh = ws + off; off += (size_t)KV_ * C_;
  _Float16* Woh = ws + off; off += (size_t)C_  * C_;
  _Float16* Qh  = ws + off; off += (size_t)M_  * C_;
  _Float16* Kh  = ws + off; off += (size_t)M_  * KV_;
  _Float16* Vth = ws + off; off += (size_t)KV_ * M_;
  _Float16* Ah  = ws + off; off += (size_t)M_  * C_;   // total ~192 MiB

  auto cvt = [&](const float* src, _Float16* dst, size_t n) {
    int n4 = (int)(n / 4);
    cvt_kernel<<<(n4 + 255) / 256, 256, 0, stream>>>(src, dst, n4);
  };
  cvt(x,  xh,  (size_t)M_  * C_);
  cvt(Wq, Wqh, (size_t)C_  * C_);
  cvt(Wk, Wkh, (size_t)KV_ * C_);
  cvt(Wv, Wvh, (size_t)KV_ * C_);
  cvt(Wo, Woh, (size_t)C_  * C_);

  dim3 blk(256);
  gemm_xwt_kernel<1><<<dim3(C_  / 128, M_ / 128), blk, 0, stream>>>(xh, Wqh, Qh,  M_, C_,  C_);
  gemm_xwt_kernel<1><<<dim3(KV_ / 128, M_ / 128), blk, 0, stream>>>(xh, Wkh, Kh,  M_, KV_, C_);
  gemm_xwt_kernel<2><<<dim3(KV_ / 128, M_ / 128), blk, 0, stream>>>(xh, Wvh, Vth, M_, KV_, C_);
  attn_kernel<<<dim3(T_ / 128, HQ_, B_), blk, 0, stream>>>(Qh, Kh, Vth, Ah);
  gemm_xwt_kernel<0><<<dim3(C_ / 128, M_ / 128), blk, 0, stream>>>(Ah, Woh, out, M_, C_, C_);
}
